// QuantLinear_13065290515021
// MI455X (gfx1250) — matrix operand order repaired and checked by reading, not compiled, not run on hardware
//
#include <hip/hip_runtime.h>
#include <stdint.h>

typedef _Float16 half_t;
typedef _Float16 v16h __attribute__((ext_vector_type(16)));
typedef _Float16 v8h  __attribute__((ext_vector_type(8)));
typedef _Float16 h2   __attribute__((ext_vector_type(2)));
typedef float    v8f  __attribute__((ext_vector_type(8)));
typedef float    v4f  __attribute__((ext_vector_type(4)));

#define BM 128
#define BN 128
#define BK 64
#define NT 128    // 128 threads = 4 wave32 waves, 2x2 wave grid of 64x64 tiles
#define LDST 72   // row stride in halves: 144 B, multiple of 16 -> aligned b128 LDS ops

// Pass 1 (when workspace available): x f32 -> f16, so the GEMM can stream A-tiles
// straight into LDS with global_load_async_to_lds_b128 (no VALU, no VGPR staging).
__global__ __launch_bounds__(256)
void x_to_f16_kernel(const float* __restrict__ x, half_t* __restrict__ xh)
{
    const size_t i = ((size_t)blockIdx.x * 256 + threadIdx.x) * 8;
    v4f a = *(const v4f*)(x + i);
    v4f b = *(const v4f*)(x + i + 4);
    v8h h;
#pragma unroll
    for (int j = 0; j < 4; ++j) { h[j] = (half_t)a[j]; h[4 + j] = (half_t)b[j]; }
    *(v8h*)(xh + i) = h;
}

// W4 (GPTQ) dequant + GEMM via v_wmma_f32_16x16x32_f16, wave32.
// Tile 128x128x64, 4 waves, each wave owns a 64x64 subtile = 4x4 WMMAs/k-step:
// every A and B fragment is reused 4x -> 1.0 ds_load_b128 per WMMA.
// Double-buffered LDS, one barrier per stage.
// XF16=true : A-tiles DMA'd global->LDS via async-to-LDS (ASYNCcnt / s_wait_asynccnt).
// XF16=false: A-tiles staged through VGPRs with f32->f16 conversion (no-workspace path).
// Dequant: nibble OR'd into f16 0x6400 mantissa -> (1024+q); v_pk_fma_f16 applies
// scale and offset -(1024+z)*scale, two weights per instruction.
template<bool XF16>
__global__ __launch_bounds__(NT)
void quantlinear_wmma_kernel(const void* __restrict__ xin,
                             const int*   __restrict__ qweight,
                             const int*   __restrict__ qzeros,
                             const float* __restrict__ scales,
                             const float* __restrict__ bias,
                             float* __restrict__ out,
                             int M, int K, int N)
{
    __shared__ half_t As[2][BM * LDST];   // As[buf][m][k] : M-major, K contiguous (f16 of x)
    __shared__ half_t Bs[2][BN * LDST];   // Bs[buf][n][k] : N-major, K contiguous (dequant W)

    const int tid    = threadIdx.x;
    const int wave   = tid >> 5;
    const int lane   = tid & 31;
    const int halfId = lane >> 4;   // K-half select per ISA A/B fragment layout
    const int l16    = lane & 15;

    const int bm = blockIdx.y * BM;
    const int bn = blockIdx.x * BN;

    const int wm = (wave & 1) * 64;   // 2 waves along M
    const int wn = (wave >> 1) * 64;  // 2 waves along N

    v8f acc[4][4];
#pragma unroll
    for (int mi = 0; mi < 4; ++mi)
#pragma unroll
        for (int ni = 0; ni < 4; ++ni)
            acc[mi][ni] = (v8f){0.f,0.f,0.f,0.f,0.f,0.f,0.f,0.f};

    // ---- weight prefetch registers: 8x128 int32 tile / 128 thr = 8 each ----
    int wr[8];
    h2  s2[8];    // packed {scale, scale}
    h2  o2[8];    // packed {-(1024+z)*scale, ...}

    auto load_w = [&](int kBase) {
        const int g = kBase >> 7;          // GROUPSIZE = 128
#pragma unroll
        for (int i = 0; i < 8; ++i) {
            const int idx = tid + i * NT;
            const int r   = idx >> 7;        // packed-K row within stage (0..7)
            const int c   = idx & 127;       // column within N tile
            const int n   = bn + c;
            wr[i] = qweight[(size_t)((kBase >> 3) + r) * N + n];
            const int qz = qzeros[(size_t)g * (N >> 3) + (n >> 3)];
            const float z = (float)(((qz >> ((n & 7) * 4)) & 15) + 1);
            const float s = scales[(size_t)g * N + n];
            const half_t sh = (half_t)s;
            const half_t oh = (half_t)(-(1024.0f + z) * s);
            s2[i] = (h2){sh, sh};
            o2[i] = (h2){oh, oh};
        }
    };

    auto store_w = [&](int buf) {
#pragma unroll
        for (int i = 0; i < 8; ++i) {
            const int idx = tid + i * NT;
            const int r   = idx >> 7;
            const int c   = idx & 127;
            const int w   = wr[i];
            union { v8h v; h2 p[4]; } U;
#pragma unroll
            for (int j = 0; j < 4; ++j) {
                // pair j = nibbles 2j (-> f16 lo) and 2j+1 (-> f16 hi)
                const int hiPart = (j < 2) ? (w << (12 - 8 * j)) : (int)((unsigned)w >> (8 * j - 12));
                const int u = ((w >> (8 * j)) & 0xF) | (hiPart & 0xF0000) | 0x64006400;
                U.p[j] = __builtin_bit_cast(h2, u) * s2[i] + o2[i];  // v_pk_fma_f16
            }
            *(v8h*)(&Bs[buf][c * LDST + r * 8]) = U.v; // 16B store, aligned (144*c + 16*r)
        }
    };

    // XF16: DMA 128x64 f16 A-tile straight into LDS (8 x 16B per thread, ASYNCcnt)
    auto async_x = [&](int kBase, int buf) {
        const half_t* xh = (const half_t*)xin;
#pragma unroll
        for (int i = 0; i < 8; ++i) {
            const int idx = tid + i * NT;    // 1024 16B chunks
            const int m   = idx >> 3;
            const int k   = (idx & 7) * 8;
            const half_t* gp = xh + (size_t)(bm + m) * K + kBase + k;
            const uint32_t lp = (uint32_t)(size_t)(&As[buf][m * LDST + k]); // low 32b of flat = LDS offset
            asm volatile("global_load_async_to_lds_b128 %0, %1, off"
                         :: "v"(lp), "v"((uint64_t)(size_t)gp) : "memory");
        }
    };

    // !XF16: direct load -> convert -> LDS store (simple fallback staging)
    auto stage_x_f32 = [&](int kBase, int buf) {
        const float* xf = (const float*)xin;
#pragma unroll
        for (int i = 0; i < 8; ++i) {
            const int idx = tid + i * NT;
            const int m   = idx >> 3;
            const int k   = (idx & 7) * 8;
            const float* p = xf + (size_t)(bm + m) * K + kBase + k;
            v4f a = *(const v4f*)p;
            v4f b = *(const v4f*)(p + 4);
            v8h h;
#pragma unroll
            for (int j = 0; j < 4; ++j) { h[j] = (half_t)a[j]; h[4 + j] = (half_t)b[j]; }
            *(v8h*)(&As[buf][m * LDST + k]) = h;
        }
    };

    auto compute = [&](int buf) {
#pragma unroll
        for (int kk = 0; kk < BK; kk += 32) {
            v16h a[4], b[4];
            // A fragment (16-bit 16x32): lanes 0-15 -> K[0..7],[16..23]; lanes 16-31 -> K[8..15],[24..31]
#pragma unroll
            for (int mi = 0; mi < 4; ++mi) {
                const half_t* p = &As[buf][(wm + mi * 16 + l16) * LDST + kk + halfId * 8];
                v8h lo = *(const v8h*)p;
                v8h hi = *(const v8h*)(p + 16);
                a[mi] = __builtin_shufflevector(lo, hi, 0,1,2,3,4,5,6,7,8,9,10,11,12,13,14,15);
            }
            // B fragment (32x16): lanes 0-15 -> K[0..7],[16..23]; lanes 16-31 -> K[8..15],[24..31]
#pragma unroll
            for (int ni = 0; ni < 4; ++ni) {
                const half_t* p = &Bs[buf][(wn + ni * 16 + l16) * LDST + kk + halfId * 8];
                v8h lo = *(const v8h*)p;
                v8h hi = *(const v8h*)(p + 16);
                b[ni] = __builtin_shufflevector(lo, hi, 0,1,2,3,4,5,6,7,8,9,10,11,12,13,14,15);
            }
#pragma unroll
            for (int mi = 0; mi < 4; ++mi)
#pragma unroll
                for (int ni = 0; ni < 4; ++ni)
                    acc[mi][ni] = __builtin_amdgcn_wmma_f32_16x16x32_f16(
                        false, a[mi], false, b[ni], (short)0, acc[mi][ni], false, false);
        }
    };

    const int steps = K / BK;

    if constexpr (XF16) {
        load_w(0);
        async_x(0, 0);
        store_w(0);
        if (steps > 1) load_w(BK);
        asm volatile("s_wait_asynccnt 0x0" ::: "memory");
        __syncthreads();
        for (int t = 0; t < steps; ++t) {
            const int cur = t & 1;
            if (t + 1 < steps) {
                async_x((t + 1) * BK, cur ^ 1);   // DMA next A-tile into other buffer
                store_w(cur ^ 1);                 // dequant next W-tile into other buffer
            }
            if (t + 2 < steps) load_w((t + 2) * BK);
            compute(cur);                          // WMMAs overlap DMA + stores
            asm volatile("s_wait_asynccnt 0x0" ::: "memory");
            __syncthreads();                       // single barrier per stage
        }
    } else {
        load_w(0);
        stage_x_f32(0, 0);
        store_w(0);
        if (steps > 1) load_w(BK);
        __syncthreads();
        for (int t = 0; t < steps; ++t) {
            const int cur = t & 1;
            if (t + 1 < steps) { stage_x_f32((t + 1) * BK, cur ^ 1); store_w(cur ^ 1); }
            if (t + 2 < steps) load_w((t + 2) * BK);
            compute(cur);
            __syncthreads();
        }
    }

    // Epilogue: C/D layout -> VGPR r holds row r (lanes 0-15) / row r+8 (lanes 16-31), lane = column
#pragma unroll
    for (int ni = 0; ni < 4; ++ni) {
        const int n  = bn + wn + ni * 16 + l16;
        const float bv = bias[n];
#pragma unroll
        for (int mi = 0; mi < 4; ++mi) {
#pragma unroll
            for (int r = 0; r < 8; ++r) {
                const int m = bm + wm + mi * 16 + halfId * 8 + r;
                out[(size_t)m * N + n] = acc[mi][ni][r] + bv;
            }
        }
    }
}

extern "C" void kernel_launch(void* const* d_in, const int* in_sizes, int n_in,
                              void* d_out, int out_size, void* d_ws, size_t ws_size,
                              hipStream_t stream) {
    const float* x       = (const float*)d_in[0];
    const int*   qweight = (const int*)  d_in[1];
    const int*   qzeros  = (const int*)  d_in[2];
    const float* scales  = (const float*)d_in[3];
    // d_in[4] = g_idx : trivially k / 128 for this model; folded into addressing
    const float* bias    = (const float*)d_in[5];
    float* out = (float*)d_out;

    const int K = in_sizes[4];            // g_idx has K entries
    const int N = in_sizes[5];            // bias has N entries
    const int M = in_sizes[0] / K;        // x is M*K

    dim3 grid(N / BN, M / BM);
    dim3 block(NT);

    const size_t need = (size_t)M * K * sizeof(half_t);
    if (ws_size >= need) {
        half_t* xh = (half_t*)d_ws;
        const int total = M * K;                       // divisible by 2048
        x_to_f16_kernel<<<total / (256 * 8), 256, 0, stream>>>(x, xh);
        quantlinear_wmma_kernel<true><<<grid, block, 0, stream>>>(
            xh, qweight, qzeros, scales, bias, out, M, K, N);
    } else {
        quantlinear_wmma_kernel<false><<<grid, block, 0, stream>>>(
            x, qweight, qzeros, scales, bias, out, M, K, N);
    }
}
